// RobotVLA_24395414241418
// MI455X (gfx1250) — matrix operand order repaired and checked by reading, not compiled, not run on hardware
//
#include <hip/hip_runtime.h>
#include <math.h>

// ---------------- constants (match reference) ----------------
#define BB     2048
#define LTXT   24
#define CS     20
#define DD     128
#define HH     4
#define EE     6
#define NLB    4
#define NLD    3
#define DFF    512
#define SDIM   16
#define JD     7

// ---------------- WMMA types ----------------
typedef __attribute__((ext_vector_type(16))) __bf16 v16bf;
typedef __attribute__((ext_vector_type(8)))  unsigned short v8us;
typedef __attribute__((ext_vector_type(8)))  float v8f;
typedef int v4i __attribute__((vector_size(16)));

union FragU { v16bf v; v8us h[2]; };

static __device__ __forceinline__ unsigned short f2bf(float f) {
    union { float f; unsigned int u; } c; c.f = f;
    unsigned int u = c.u;
    unsigned int r = (u + 0x7fffu + ((u >> 16) & 1u)) >> 16;
    return (unsigned short)r;
}

// CDNA5 async global->LDS path (guarded; falls back to explicit b128 copy)
#if defined(__HIP_DEVICE_COMPILE__) && defined(__gfx1250__) && \
    __has_builtin(__builtin_amdgcn_global_load_async_to_lds_b128) && \
    __has_builtin(__builtin_amdgcn_s_wait_asynccnt)
#define USE_ASYNC_LDS 1
#else
#define USE_ASYNC_LDS 0
#endif

// =====================================================================
// Weight pre-conversion: W[K,N] f32  ->  Wt[N,KT] bf16 (transposed,
// K zero-padded to KT). `count` matrices back to back.
// =====================================================================
__global__ void cvt_w_kernel(const float* __restrict__ W, unsigned short* __restrict__ Wt,
                             int K, int N, int KT, long total)
{
    long i = (long)blockIdx.x * 256 + threadIdx.x;
    if (i >= total) return;
    long per = (long)N * KT;
    long m = i / per;
    long r = i - m * per;
    int n = (int)(r / KT);
    int k = (int)(r - (long)n * KT);
    unsigned short v = 0;
    if (k < K) v = f2bf(W[m * (long)K * N + (long)k * N + n]);
    Wt[i] = v;
}

// =====================================================================
// GEMM: C[M,N] = act(A[M,K] @ W[K,N] + bias[N])
// A: f32 [M,K] (lda). W pre-converted/transposed: bf16 [N,KT].
// Block: 256 thr (8 waves). Block tile 256x64; wave tile 32x64 (8 WMMA/step).
// M % 256 == 0, N % 64 == 0, K/KT compile-time.
// ACT: 0 none, 1 gelu(exact), 2 relu
// =====================================================================
template<int ACT, int K, int KT>
__global__ __launch_bounds__(256) void gemm_wmma_kernel(
    const float* __restrict__ A, int lda,
    const unsigned short* __restrict__ Wt,
    const float* __restrict__ bias,
    float* __restrict__ C, int ldc, int N)
{
    __shared__ __align__(16) unsigned short lA[256 * 32];  // [m][k]
    __shared__ __align__(16) unsigned short lB[64 * 32];   // [n][k]

    const int tid  = threadIdx.x;
    const int lane = tid & 31;
    const int wave = tid >> 5;
    const int m0   = blockIdx.y * 256;
    const int n0   = blockIdx.x * 64;

    v8f acc[2][4] = {};

    for (int k0 = 0; k0 < KT; k0 += 32) {
        // ---- stage A tile 256x32: 8 float4 chunks per thread ----
        #pragma unroll
        for (int i = 0; i < 8; ++i) {
            const int c  = tid + i * 256;
            const int ml = c >> 3;
            const int kl = (c & 7) * 4;
            uint2 packed;
            if (K == KT || kl < K) {   // compile-time for full tiles; K=16 pads
                const float4 f = *reinterpret_cast<const float4*>(
                    &A[(size_t)(m0 + ml) * lda + k0 + kl]);
                packed.x = (unsigned)f2bf(f.x) | ((unsigned)f2bf(f.y) << 16);
                packed.y = (unsigned)f2bf(f.z) | ((unsigned)f2bf(f.w) << 16);
            } else {
                packed.x = 0u; packed.y = 0u;
            }
            *reinterpret_cast<uint2*>(&lA[ml * 32 + kl]) = packed;
            if (i == 0 && k0 + 32 < KT)
                __builtin_prefetch(&A[(size_t)(m0 + ml) * lda + k0 + 32 + kl], 0, 0);
        }
        // ---- stage B tile 64x32 bf16: one 16B chunk per thread ----
        {
            const int nl = tid >> 2;
            const int kc = (tid & 3) * 8;
            const unsigned short* src = &Wt[(size_t)(n0 + nl) * KT + k0 + kc];
#if USE_ASYNC_LDS
            __builtin_amdgcn_global_load_async_to_lds_b128(
                (__attribute__((address_space(1))) v4i*)src,
                (__attribute__((address_space(3))) v4i*)&lB[nl * 32 + kc],
                0, 0);
#else
            *reinterpret_cast<v8us*>(&lB[nl * 32 + kc]) =
                *reinterpret_cast<const v8us*>(src);
#endif
        }
#if USE_ASYNC_LDS
        __builtin_amdgcn_s_wait_asynccnt(0);
#endif
        __syncthreads();

        // ---- fragments ----
        FragU af0, af1;
        const int mr = wave * 32 + (lane & 15);
        const int kh = (lane >> 4) * 8;
        af0.h[0] = *(const v8us*)&lA[mr * 32 + kh];
        af0.h[1] = *(const v8us*)&lA[mr * 32 + 16 + kh];
        af1.h[0] = *(const v8us*)&lA[(mr + 16) * 32 + kh];
        af1.h[1] = *(const v8us*)&lA[(mr + 16) * 32 + 16 + kh];

        const int kb = (lane >> 4) * 8;
        #pragma unroll
        for (int nt = 0; nt < 4; ++nt) {
            FragU bfr;
            const int ncol = nt * 16 + (lane & 15);
            bfr.h[0] = *(const v8us*)&lB[ncol * 32 + kb];
            bfr.h[1] = *(const v8us*)&lB[ncol * 32 + 16 + kb];
            acc[0][nt] = __builtin_amdgcn_wmma_f32_16x16x32_bf16(
                             false, af0.v, false, bfr.v, (short)0, acc[0][nt], false, false);
            acc[1][nt] = __builtin_amdgcn_wmma_f32_16x16x32_bf16(
                             false, af1.v, false, bfr.v, (short)0, acc[1][nt], false, false);
        }
        __syncthreads();
    }

    // ---- epilogue: C/D layout: vgpr r -> m = r + (lane>=16 ? 8:0), n = lane&15
    const int nl = lane & 15;
    #pragma unroll
    for (int mt = 0; mt < 2; ++mt) {
        const int mbase = m0 + wave * 32 + mt * 16 + ((lane >> 4) ? 8 : 0);
        #pragma unroll
        for (int nt = 0; nt < 4; ++nt) {
            const int n = n0 + nt * 16 + nl;
            const float bv = bias ? bias[n] : 0.0f;
            #pragma unroll
            for (int r = 0; r < 8; ++r) {
                float v = acc[mt][nt][r] + bv;
                if (ACT == 1) v = 0.5f * v * (1.0f + erff(v * 0.70710678118654752f));
                else if (ACT == 2) v = fmaxf(v, 0.0f);
                C[(size_t)(mbase + r) * ldc + n] = v;
            }
        }
    }
}

// =====================================================================
// LayerNorm (D=128), optional residual: Y = LN(X + R) * s + b
// =====================================================================
__global__ __launch_bounds__(128) void ln_kernel(
    const float* __restrict__ X, long ldx,
    const float* __restrict__ R,
    const float* __restrict__ s, const float* __restrict__ b,
    float* __restrict__ Y, long ldy)
{
    __shared__ float red[128];
    const int row = blockIdx.x;
    const int t   = threadIdx.x;
    float v = X[(size_t)row * ldx + t];
    if (R) v += R[(size_t)row * 128 + t];
    red[t] = v; __syncthreads();
    #pragma unroll
    for (int o = 64; o > 0; o >>= 1) { if (t < o) red[t] += red[t + o]; __syncthreads(); }
    const float mean = red[0] * (1.0f / 128.0f);
    __syncthreads();
    const float d = v - mean;
    red[t] = d * d; __syncthreads();
    #pragma unroll
    for (int o = 64; o > 0; o >>= 1) { if (t < o) red[t] += red[t + o]; __syncthreads(); }
    const float var = red[0] * (1.0f / 128.0f);
    Y[(size_t)row * ldy + t] = d * rsqrtf(var + 1e-5f) * s[t] + b[t];
}

// =====================================================================
// Attention softmax-context per (batch, head); head dim 32, 4 heads.
// =====================================================================
template<int LQ, int LK, bool MASK>
__global__ __launch_bounds__(32) void attn_kernel(
    const float* __restrict__ Q, int ldq,
    const float* __restrict__ KV, int ldkv, int koff, int voff,
    float* __restrict__ ctx,
    const int* __restrict__ pad_ids)
{
    __shared__ float kl[LK * 32], vl[LK * 32], mk[LK];
    const int b = blockIdx.x >> 2;
    const int h = blockIdx.x & 3;
    const int t = threadIdx.x;

    for (int e = t; e < LK * 32; e += 32) {
        int kk = e >> 5, d = e & 31;
        size_t r = (size_t)(b * LK + kk) * ldkv + h * 32;
        kl[e] = KV[r + koff + d];
        vl[e] = KV[r + voff + d];
    }
    if (t < LK) mk[t] = (MASK && pad_ids[b * LK + t] == 0) ? -1e9f : 0.0f;
    __syncthreads();

    if (t < LQ) {
        float q[32];
        const size_t qr = (size_t)(b * LQ + t) * ldq + h * 32;
        #pragma unroll
        for (int d = 0; d < 32; ++d) q[d] = Q[qr + d];
        float sc[LK];
        float mx = -1e30f;
        #pragma unroll
        for (int kk = 0; kk < LK; ++kk) {
            float a = 0.0f;
            #pragma unroll
            for (int d = 0; d < 32; ++d) a += q[d] * kl[kk * 32 + d];
            a = a * 0.17677669529663687f + mk[kk];
            sc[kk] = a; mx = fmaxf(mx, a);
        }
        float sum = 0.0f;
        #pragma unroll
        for (int kk = 0; kk < LK; ++kk) { sc[kk] = __expf(sc[kk] - mx); sum += sc[kk]; }
        const float inv = 1.0f / sum;
        float o[32];
        #pragma unroll
        for (int d = 0; d < 32; ++d) o[d] = 0.0f;
        #pragma unroll
        for (int kk = 0; kk < LK; ++kk) {
            const float w = sc[kk];
            #pragma unroll
            for (int d = 0; d < 32; ++d) o[d] += w * vl[kk * 32 + d];
        }
        const size_t orow = (size_t)(b * LQ + t) * 128 + h * 32;
        #pragma unroll
        for (int d = 0; d < 32; ++d) ctx[orow + d] = o[d] * inv;
    }
}

// ---------------- small element-wise kernels ----------------
__global__ void embed_kernel(const int* __restrict__ ids, const float* __restrict__ tok,
                             const float* __restrict__ pos, float* __restrict__ x, int total)
{
    int i = blockIdx.x * 256 + threadIdx.x;
    if (i >= total) return;
    int d  = i & 127;
    int bt = i >> 7;
    int t  = bt % LTXT;
    int id = ids[bt];
    x[i] = tok[id * 128 + d] + pos[t * 128 + d];
}

__global__ void add_kernel(float* __restrict__ x, const float* __restrict__ r, int n)
{
    int i = blockIdx.x * 256 + threadIdx.x;
    if (i < n) x[i] += r[i];
}

__global__ void memrow1_kernel(float* __restrict__ mem, const float* __restrict__ zb, int n)
{
    int i = blockIdx.x * 256 + threadIdx.x;
    if (i >= n) return;
    int b = i >> 7, d = i & 127;
    mem[(size_t)b * 256 + 128 + d] = zb[d];
}

__global__ void xqinit_kernel(float* __restrict__ xq, const float* __restrict__ qpe, int n)
{
    int i = blockIdx.x * 256 + threadIdx.x;
    if (i >= n) return;
    int d = i & 127;
    int t = (i >> 7) % CS;
    xq[i] = qpe[t * 128 + d];
}

__global__ __launch_bounds__(128) void pe_kernel(float* __restrict__ qpe)
{
    int t = blockIdx.x, d = threadIdx.x;
    int i = d >> 1;
    float div = __expf(-logf(10000.0f) * (float)(2 * i) * (1.0f / 128.0f));
    float ang = (float)t * div;
    qpe[t * 128 + d] = (d & 1) ? cosf(ang) : sinf(ang);
}

__global__ void gate_kernel(const float* __restrict__ fused, const float* __restrict__ gW,
                            const float* __restrict__ gb, float* __restrict__ wfull)
{
    int b = blockIdx.x * 64 + threadIdx.x;
    if (b >= BB) return;
    float lg[EE];
    #pragma unroll
    for (int e = 0; e < EE; ++e) lg[e] = gb[e];
    for (int k = 0; k < 128; ++k) {
        float f = fused[(size_t)b * 128 + k];
        #pragma unroll
        for (int e = 0; e < EE; ++e) lg[e] += f * gW[k * EE + e];
    }
    int i1 = 0;
    #pragma unroll
    for (int e = 1; e < EE; ++e) if (lg[e] > lg[i1]) i1 = e;
    int i2 = -1;
    #pragma unroll
    for (int e = 0; e < EE; ++e) if (e != i1 && (i2 < 0 || lg[e] > lg[i2])) i2 = e;
    float e2 = __expf(lg[i2] - lg[i1]);
    float inv = 1.0f / (1.0f + e2);
    #pragma unroll
    for (int e = 0; e < EE; ++e) wfull[(size_t)b * EE + e] = 0.0f;
    wfull[(size_t)b * EE + i1] = inv;
    wfull[(size_t)b * EE + i2] = e2 * inv;
}

__global__ __launch_bounds__(32) void head_kernel(
    const float* __restrict__ xq, const float* __restrict__ hW, const float* __restrict__ hb,
    const float* __restrict__ wfull, int e, float* __restrict__ out)
{
    int row = blockIdx.x;
    int j = threadIdx.x;
    if (j >= JD) return;
    int b = row / CS;
    float acc = hb[j];
    #pragma unroll
    for (int k = 0; k < 128; ++k) acc += xq[(size_t)row * 128 + k] * hW[k * JD + j];
    out[(size_t)row * JD + j] += wfull[(size_t)b * EE + e] * acc;
}

// ---------------- host-side helpers ----------------
template<int ACT, int K, int KT>
static inline void gemm_t(const float* A, int lda, const unsigned short* Wt,
                          const float* bias, float* C, int ldc, int M, int N, hipStream_t s)
{
    dim3 g(N / 64, M / 256), b(256);
    gemm_wmma_kernel<ACT, K, KT><<<g, b, 0, s>>>(A, lda, Wt, bias, C, ldc, N);
}

static inline void cvt(const float* W, unsigned short* Wt, int K, int N, int KT,
                       int count, hipStream_t s)
{
    long total = (long)count * N * KT;
    cvt_w_kernel<<<(int)((total + 255) / 256), 256, 0, s>>>(W, Wt, K, N, KT, total);
}

extern "C" void kernel_launch(void* const* d_in, const int* in_sizes, int n_in,
                              void* d_out, int out_size, void* d_ws, size_t ws_size,
                              hipStream_t stream)
{
    (void)in_sizes; (void)n_in; (void)out_size; (void)ws_size;

    // ---- inputs (setup_inputs order) ----
    const int*   ids       = (const int*)  d_in[0];
    const float* state     = (const float*)d_in[1];
    const float* tok_emb   = (const float*)d_in[2];
    const float* pos_emb   = (const float*)d_in[3];
    const float* bWqkv     = (const float*)d_in[4];
    const float* bbqkv     = (const float*)d_in[5];
    const float* bWo       = (const float*)d_in[6];
    const float* bbo       = (const float*)d_in[7];
    const float* bW1       = (const float*)d_in[8];
    const float* bb1       = (const float*)d_in[9];
    const float* bW2       = (const float*)d_in[10];
    const float* bb2       = (const float*)d_in[11];
    const float* bln_s     = (const float*)d_in[12];
    const float* bln_b     = (const float*)d_in[13];
    const float* lnf_s     = (const float*)d_in[14];
    const float* lnf_b     = (const float*)d_in[15];
    const float* se_W1     = (const float*)d_in[16];
    const float* se_b1     = (const float*)d_in[17];
    const float* se_W2     = (const float*)d_in[18];
    const float* se_b2     = (const float*)d_in[19];
    const float* se_ln_s   = (const float*)d_in[20];
    const float* se_ln_b   = (const float*)d_in[21];
    const float* r_W1      = (const float*)d_in[22];
    const float* r_b1      = (const float*)d_in[23];
    const float* r_W2      = (const float*)d_in[24];
    const float* r_b2      = (const float*)d_in[25];
    const float* r_ln_s    = (const float*)d_in[26];
    const float* r_ln_b    = (const float*)d_in[27];
    const float* gate_W    = (const float*)d_in[28];
    const float* gate_b    = (const float*)d_in[29];
    const float* sp_W      = (const float*)d_in[30];
    const float* sp_b      = (const float*)d_in[31];
    /* exp_zp_W (d_in[32]) unused: z == 0 */
    const float* zp_b      = (const float*)d_in[33];
    const float* aWqkv     = (const float*)d_in[34];
    const float* abqkv     = (const float*)d_in[35];
    const float* aWo       = (const float*)d_in[36];
    const float* abo       = (const float*)d_in[37];
    const float* eln_s     = (const float*)d_in[38];
    const float* eln_b     = (const float*)d_in[39];
    const float* fW1       = (const float*)d_in[40];
    const float* fb1       = (const float*)d_in[41];
    const float* fW2       = (const float*)d_in[42];
    const float* fb2       = (const float*)d_in[43];
    const float* hW        = (const float*)d_in[44];
    const float* hb        = (const float*)d_in[45];

    float* out = (float*)d_out;

    // ---- workspace carving ----
    char* base = (char*)d_ws;
    size_t off = 0;
    auto allocf = [&](size_t n) -> float* {
        float* p = (float*)(base + off);
        off += ((n * sizeof(float)) + 255) & ~(size_t)255;
        return p;
    };
    auto allocu = [&](size_t n) -> unsigned short* {
        unsigned short* p = (unsigned short*)(base + off);
        off += ((n * sizeof(unsigned short)) + 255) & ~(size_t)255;
        return p;
    };
    const int M1 = BB * LTXT;    // 49152
    const int M2 = BB * CS;      // 40960
    float* x     = allocf((size_t)M1 * 128);   // BERT acts; aliased as expert xq
    float* big   = allocf((size_t)M1 * 512);   // qkv / ffn-hidden / cross-kv
    float* bufA  = allocf((size_t)M1 * 128);   // attention context
    float* bufB  = allocf((size_t)M1 * 128);   // projection outputs
    float* xn    = allocf((size_t)M2 * 128);
    float* memb  = allocf((size_t)BB * 256);
    float* fcat  = allocf((size_t)BB * 256);
    float* fused = allocf((size_t)BB * 128);
    float* wfull = allocf((size_t)BB * EE + 64);
    float* qpe   = allocf(CS * 128);
    float* xq    = x;

    // bf16 transposed weights [N][KT]
    unsigned short* t_bqkv = allocu((size_t)NLB * 384 * 128);
    unsigned short* t_bWo  = allocu((size_t)NLB * 128 * 128);
    unsigned short* t_bW1  = allocu((size_t)NLB * 512 * 128);
    unsigned short* t_bW2  = allocu((size_t)NLB * 128 * 512);
    unsigned short* t_se1  = allocu((size_t)128 * 32);
    unsigned short* t_se2  = allocu((size_t)128 * 128);
    unsigned short* t_r1   = allocu((size_t)128 * 256);
    unsigned short* t_r2   = allocu((size_t)128 * 128);
    unsigned short* t_sp   = allocu((size_t)EE * 128 * 32);
    unsigned short* t_aqkv = allocu((size_t)EE * NLD * 2 * 384 * 128);
    unsigned short* t_aWo  = allocu((size_t)EE * NLD * 2 * 128 * 128);
    unsigned short* t_fW1  = allocu((size_t)EE * NLD * 512 * 128);
    unsigned short* t_fW2  = allocu((size_t)EE * NLD * 128 * 512);

    // ---- one-time (per launch) weight conversion ----
    cvt(bWqkv, t_bqkv, 128, 384, 128, NLB, stream);
    cvt(bWo,   t_bWo,  128, 128, 128, NLB, stream);
    cvt(bW1,   t_bW1,  128, 512, 128, NLB, stream);
    cvt(bW2,   t_bW2,  512, 128, 512, NLB, stream);
    cvt(se_W1, t_se1,   16, 128,  32, 1, stream);
    cvt(se_W2, t_se2,  128, 128, 128, 1, stream);
    cvt(r_W1,  t_r1,   256, 128, 256, 1, stream);
    cvt(r_W2,  t_r2,   128, 128, 128, 1, stream);
    cvt(sp_W,  t_sp,    16, 128,  32, EE, stream);
    cvt(aWqkv, t_aqkv, 128, 384, 128, EE * NLD * 2, stream);
    cvt(aWo,   t_aWo,  128, 128, 128, EE * NLD * 2, stream);
    cvt(fW1,   t_fW1,  128, 512, 128, EE * NLD, stream);
    cvt(fW2,   t_fW2,  512, 128, 512, EE * NLD, stream);

    // ================= MiniBERT =================
    embed_kernel<<<(M1 * 128 + 255) / 256, 256, 0, stream>>>(ids, tok_emb, pos_emb, x, M1 * 128);
    for (int l = 0; l < NLB; ++l) {
        gemm_t<0,128,128>(x, 128, t_bqkv + (size_t)l * 384 * 128, bbqkv + l * 384, big, 384, M1, 384, stream);
        attn_kernel<LTXT, LTXT, true><<<BB * HH, 32, 0, stream>>>(big, 384, big, 384, 128, 256, bufA, ids);
        gemm_t<0,128,128>(bufA, 128, t_bWo + (size_t)l * 128 * 128, bbo + l * 128, bufB, 128, M1, 128, stream);
        ln_kernel<<<M1, 128, 0, stream>>>(x, 128, bufB, bln_s + (l * 2 + 0) * 128, bln_b + (l * 2 + 0) * 128, x, 128);
        gemm_t<1,128,128>(x, 128, t_bW1 + (size_t)l * 512 * 128, bb1 + l * 512, big, 512, M1, 512, stream);
        gemm_t<0,512,512>(big, 512, t_bW2 + (size_t)l * 128 * 512, bb2 + l * 128, bufB, 128, M1, 128, stream);
        ln_kernel<<<M1, 128, 0, stream>>>(x, 128, bufB, bln_s + (l * 2 + 1) * 128, bln_b + (l * 2 + 1) * 128, x, 128);
    }
    ln_kernel<<<BB, 128, 0, stream>>>(x, (long)LTXT * 128, nullptr, lnf_s, lnf_b, fcat, 256);

    // ================= StateEncoder =================
    gemm_t<1,16,32>(state, SDIM, t_se1, se_b1, bufA, 128, BB, 128, stream);
    gemm_t<0,128,128>(bufA, 128, t_se2, se_b2, bufB, 128, BB, 128, stream);
    ln_kernel<<<BB, 128, 0, stream>>>(bufB, 128, nullptr, se_ln_s, se_ln_b, fcat + 128, 256);

    // ================= FusionRouter =================
    gemm_t<1,256,256>(fcat, 256, t_r1, r_b1, bufA, 128, BB, 128, stream);
    gemm_t<0,128,128>(bufA, 128, t_r2, r_b2, bufB, 128, BB, 128, stream);
    ln_kernel<<<BB, 128, 0, stream>>>(bufB, 128, nullptr, r_ln_s, r_ln_b, fused, 128);
    gate_kernel<<<(BB + 63) / 64, 64, 0, stream>>>(fused, gate_W, gate_b, wfull);

    // ================= ACT Experts =================
    pe_kernel<<<CS, 128, 0, stream>>>(qpe);
    (void)hipMemsetAsync(out, 0, (size_t)M2 * JD * sizeof(float), stream);

    for (int e = 0; e < EE; ++e) {
        gemm_t<0,16,32>(state, SDIM, t_sp + (size_t)e * 128 * 32, sp_b + e * 128, memb, 256, BB, 128, stream);
        memrow1_kernel<<<(BB * 128 + 255) / 256, 256, 0, stream>>>(memb, zp_b + e * 128, BB * 128);
        xqinit_kernel<<<(M2 * 128 + 255) / 256, 256, 0, stream>>>(xq, qpe, M2 * 128);

        for (int l = 0; l < NLD; ++l) {
            const size_t el = (size_t)e * NLD + l;
            const unsigned short* sW = t_aqkv + (el * 2 + 0) * 384 * 128;
            const unsigned short* cW = t_aqkv + (el * 2 + 1) * 384 * 128;  // rows 0..127: q
            const unsigned short* cWkv = cW + (size_t)128 * 128;           // rows 128..383: k|v
            const float* sB  = abqkv + (el * 2 + 0) * 384;
            const float* cB  = abqkv + (el * 2 + 1) * 384;
            const unsigned short* oWs = t_aWo + (el * 2 + 0) * 128 * 128;
            const unsigned short* oWc = t_aWo + (el * 2 + 1) * 128 * 128;
            const float* oBs = abo + (el * 2 + 0) * 128;
            const float* oBc = abo + (el * 2 + 1) * 128;
            const float* ls0 = eln_s + (el * 3 + 0) * 128; const float* lb0 = eln_b + (el * 3 + 0) * 128;
            const float* ls1 = eln_s + (el * 3 + 1) * 128; const float* lb1 = eln_b + (el * 3 + 1) * 128;
            const float* ls2 = eln_s + (el * 3 + 2) * 128; const float* lb2 = eln_b + (el * 3 + 2) * 128;

            // --- self attention (norm_first) ---
            ln_kernel<<<M2, 128, 0, stream>>>(xq, 128, nullptr, ls0, lb0, xn, 128);
            gemm_t<0,128,128>(xn, 128, sW, sB, big, 384, M2, 384, stream);
            attn_kernel<CS, CS, false><<<BB * HH, 32, 0, stream>>>(big, 384, big, 384, 128, 256, bufA, nullptr);
            gemm_t<0,128,128>(bufA, 128, oWs, oBs, bufB, 128, M2, 128, stream);
            add_kernel<<<(M2 * 128 + 255) / 256, 256, 0, stream>>>(xq, bufB, M2 * 128);

            // --- cross attention on 2-token memory ---
            ln_kernel<<<M2, 128, 0, stream>>>(xq, 128, nullptr, ls1, lb1, xn, 128);
            gemm_t<0,128,128>(xn, 128, cW, cB, bufB, 128, M2, 128, stream);              // q
            gemm_t<0,128,128>(memb, 128, cWkv, cB + 128, big, 256, BB * 2, 256, stream); // k|v
            attn_kernel<CS, 2, false><<<BB * HH, 32, 0, stream>>>(bufB, 128, big, 256, 0, 128, bufA, nullptr);
            gemm_t<0,128,128>(bufA, 128, oWc, oBc, bufB, 128, M2, 128, stream);
            add_kernel<<<(M2 * 128 + 255) / 256, 256, 0, stream>>>(xq, bufB, M2 * 128);

            // --- FFN (ReLU) ---
            ln_kernel<<<M2, 128, 0, stream>>>(xq, 128, nullptr, ls2, lb2, xn, 128);
            gemm_t<2,128,128>(xn, 128, t_fW1 + el * 512 * 128, fb1 + el * 512, big, 512, M2, 512, stream);
            gemm_t<0,512,512>(big, 512, t_fW2 + el * 128 * 512, fb2 + el * 128, bufB, 128, M2, 128, stream);
            add_kernel<<<(M2 * 128 + 255) / 256, 256, 0, stream>>>(xq, bufB, M2 * 128);
        }
        head_kernel<<<M2, 32, 0, stream>>>(xq, hW + (size_t)e * 128 * JD, hb + e * JD, wfull, e, out);
    }
}
